// DabDeformableDetrTransformer_58437325029881
// MI455X (gfx1250) — matrix operand order repaired and checked by reading, not compiled, not run on hardware
//
#include <hip/hip_runtime.h>
#include <hip/hip_bf16.h>
#include <math.h>

// ---------------------------------------------------------------------------
// Types for CDNA5 WMMA (wave32): A/B = 16 bf16 per lane, C/D = 8 f32 per lane
// ---------------------------------------------------------------------------
typedef __bf16 bf16;
typedef __attribute__((ext_vector_type(16))) __bf16 v16bf;
typedef __attribute__((ext_vector_type(8)))  __bf16 v8bf;
typedef __attribute__((ext_vector_type(8)))  float  v8f;

#define S_TOT   19947
#define BATCH   8
#define M_TOT   (BATCH * S_TOT)     // 159576
#define CDIM    256
#define KSTEPS  (CDIM / 32)         // 8
#define NCLS    91
#define NCLS_PAD 128                // padded to 8 N-tiles: no guards in GEMM
#define TOPK    20
#define NMS_THR 0.5f

// spatial levels: (100,150),(50,75),(25,38),(13,19)
__device__ __forceinline__ void level_decode(int s, int& lvl, int& y, int& x) {
    const int offs[5] = {0, 15000, 18750, 19700, 19947};
    const int Ws[4]   = {150, 75, 38, 19};
    lvl = 0;
    while (lvl < 3 && s >= offs[lvl + 1]) ++lvl;
    int loc = s - offs[lvl];
    y = loc / Ws[lvl];
    x = loc % Ws[lvl];
}

// ---------------------------------------------------------------------------
// Kernel 0: per (batch, level) valid_W / valid_H counts  (trivial work)
// ---------------------------------------------------------------------------
__global__ void validwh_kernel(const unsigned char* __restrict__ mask,
                               float* __restrict__ vwh) {
    const int offs[4] = {0, 15000, 18750, 19700};
    const int Hs[4]   = {100, 50, 25, 13};
    const int Ws[4]   = {150, 75, 38, 19};
    int b = blockIdx.x >> 2, lvl = blockIdx.x & 3;
    if (threadIdx.x != 0) return;
    const unsigned char* m = mask + (size_t)b * S_TOT + offs[lvl];
    int vh = 0, vw = 0;
    for (int r = 0; r < Hs[lvl]; ++r) vh += (m[r * Ws[lvl]] == 0);
    for (int c = 0; c < Ws[lvl]; ++c) vw += (m[c] == 0);
    vwh[(b * 4 + lvl) * 2 + 0] = (float)vw;
    vwh[(b * 4 + lvl) * 2 + 1] = (float)vh;
}

// ---------------------------------------------------------------------------
// Kernel 1: proposals (logits / +inf) + masked fp32->bf16 copy of memory.
// One wave per token (8 tokens / block).
// ---------------------------------------------------------------------------
__global__ __launch_bounds__(256) void proposals_kernel(
        const float* __restrict__ mem, const unsigned char* __restrict__ mask,
        const float* __restrict__ vwh, bf16* __restrict__ memA,
        float* __restrict__ opf) {
    int wave = threadIdx.x >> 5, lane = threadIdx.x & 31;
    int tok = blockIdx.x * 8 + wave;
    if (tok >= M_TOT) return;
    int b = tok / S_TOT, s = tok - b * S_TOT;
    int lvl, y, x;
    level_decode(s, lvl, y, x);
    float vw = vwh[(b * 4 + lvl) * 2 + 0];
    float vh = vwh[(b * 4 + lvl) * 2 + 1];
    float px = ((float)x + 0.5f) / vw;
    float py = ((float)y + 0.5f) / vh;
    float wh = 0.05f * (float)(1 << lvl);
    bool valid = (px > 0.01f) & (px < 0.99f) & (py > 0.01f) & (py < 0.99f) &
                 (wh > 0.01f) & (wh < 0.99f);
    bool invalid = (mask[(size_t)b * S_TOT + s] != 0) || !valid;
    if (lane < 4) {
        float op = (lane == 0) ? px : (lane == 1) ? py : wh;
        float safe = valid ? op : 0.5f;
        float lg = logf(safe / (1.0f - safe));
        opf[(size_t)tok * 4 + lane] = invalid ? __builtin_inff() : lg;
    }
    const float* src = mem + (size_t)tok * CDIM;
    bf16* dst = memA + (size_t)tok * CDIM;
    for (int j = lane; j < CDIM; j += 32)
        dst[j] = (bf16)(invalid ? 0.0f : src[j]);
}

// ---------------------------------------------------------------------------
// Kernel 2: pack fp32 weights [K=256][Nreal] row-major into a CDNA5 WMMA
// B-operand lane layout (contiguous k = 16h + e, same as the A loads): per (ntile, kstep): 32 lanes x 16 bf16,
// lane<16 -> K=kstep*32+0..15, N=tile*16+lane ; lane>=16 -> K=+16..31.
// One thread per (ntile, kstep, lane) -> one contiguous 32B group.
// Columns >= Nreal are zero-padded.
// ---------------------------------------------------------------------------
__global__ void packB_kernel(const float* __restrict__ W, bf16* __restrict__ Bp,
                             int Nreal, int NT) {
    int id = blockIdx.x * blockDim.x + threadIdx.x;
    int total = NT * KSTEPS * 32;
    if (id >= total) return;
    int lane  = id & 31;
    int kstep = (id >> 5) % KSTEPS;
    int t     = id / (32 * KSTEPS);
    int n     = t * 16 + (lane & 15);
    int kbase = kstep * 32 + ((lane >> 4) << 4);
    bf16* dst = Bp + (size_t)id * 16;
    for (int e = 0; e < 16; ++e) {
        float v = (n < Nreal) ? W[(size_t)(kbase + e) * Nreal + n] : 0.0f;
        dst[e] = (bf16)v;
    }
}

// ---------------------------------------------------------------------------
// Kernel 3: WMMA GEMM, A[M x 256] bf16 row-major, B packed, fused epilogue.
//   EPI 0: LayerNorm(acc + bias)*g+b  -> bf16 out
//   EPI 1: relu(acc + bias)           -> bf16 out
//   EPI 2: rowmax over 91 of acc+bias -> scores (no Out)
// Block: 32 rows x (NT*16) cols, 8 waves = 2 M-tiles x 4 N-groups.
// NT must be a multiple of 4: every wave owns exactly NT/4 tiles, so the
// tile loops are guard-free straight-line code after unrolling and the
// accumulators stay in fixed VGPRs with no exec-mask branching.
// ---------------------------------------------------------------------------
template <int NT, int EPI>
__global__ __launch_bounds__(256) void wmma_gemm(
        const bf16* __restrict__ A, const bf16* __restrict__ Bp,
        const float* __restrict__ bias, const float* __restrict__ ln_g,
        const float* __restrict__ ln_b, bf16* __restrict__ Out,
        float* __restrict__ scores, int M) {
    static_assert(NT % 4 == 0, "NT must be a multiple of 4");
    constexpr int TPW = NT / 4;         // tiles per wave (4 N-groups)
    __shared__ float lds[32 * NT * 16];
    const int N = NT * 16;
    int wave = threadIdx.x >> 5, lane = threadIdx.x & 31;
    int mt = wave & 1;            // M-tile within block
    int g  = wave >> 1;           // N-group
    int m0 = blockIdx.x * 32 + mt * 16;
    int row = m0 + (lane & 15);
    int rowc = row < M ? row : M - 1;
    int half = (lane >> 4) << 3;  // 0 or 8 (A K-half select: lane half h loads k 16h..16h+15)

    v8f acc[TPW];
#pragma unroll
    for (int i = 0; i < TPW; ++i) acc[i] = (v8f){0,0,0,0,0,0,0,0};

    const bf16* arow = A + (size_t)rowc * CDIM;
    const bf16* bbase = Bp + ((size_t)g * TPW * KSTEPS * 32 + lane) * 16;
#pragma unroll 2
    for (int ks = 0; ks < KSTEPS; ++ks) {
        const bf16* ap = arow + ks * 32;
        __builtin_prefetch(ap + 32, 0, 1);   // speculative; past-end is dropped
        union { v16bf v; v8bf h[2]; } ua;
        ua.h[0] = *reinterpret_cast<const v8bf*>(ap + 2 * half);        // K 2*half..2*half+7
        ua.h[1] = *reinterpret_cast<const v8bf*>(ap + 2 * half + 8);   // K 2*half+8..
#pragma unroll
        for (int i = 0; i < TPW; ++i) {
            const bf16* bptr = bbase + ((size_t)i * KSTEPS + ks) * 32 * 16;
            v16bf bm = *reinterpret_cast<const v16bf*>(bptr);
            acc[i] = __builtin_amdgcn_wmma_f32_16x16x32_bf16(
                false, ua.v, false, bm, (short)0, acc[i], false, false);
        }
    }

    // stage accumulators to LDS (D layout: vgpr r = row, lanes>=16 -> row+8)
    {
        int rbase = mt * 16 + ((lane >> 4) << 3);
#pragma unroll
        for (int i = 0; i < TPW; ++i) {
            int col = (g * TPW + i) * 16 + (lane & 15);
#pragma unroll
            for (int r = 0; r < 8; ++r)
                lds[(rbase + r) * N + col] = acc[i][r];
        }
    }
    __syncthreads();

    if (EPI == 0) {  // LayerNorm epilogue, one thread per row
        if (threadIdx.x < 32) {
            int r = threadIdx.x, gm = blockIdx.x * 32 + r;
            if (gm < M) {
                float mu = 0.f, sq = 0.f;
                for (int j = 0; j < N; ++j) {
                    float v = lds[r * N + j] + bias[j];
                    mu += v; sq += v * v;
                }
                mu *= (1.0f / N);
                float var = sq * (1.0f / N) - mu * mu;
                float rs = rsqrtf(var + 1e-5f);
                bf16* o = Out + (size_t)gm * N;
                for (int j = 0; j < N; ++j) {
                    float v = lds[r * N + j] + bias[j];
                    o[j] = (bf16)((v - mu) * rs * ln_g[j] + ln_b[j]);
                }
            }
        }
    } else if (EPI == 1) {  // ReLU epilogue, all threads
        for (int i = threadIdx.x; i < 32 * N; i += 256) {
            int r = i / N, c = i - r * N;
            int gm = blockIdx.x * 32 + r;
            if (gm < M) {
                float v = lds[i] + bias[c];
                Out[(size_t)gm * N + c] = (bf16)(v > 0.f ? v : 0.f);
            }
        }
    } else {  // EPI == 2: row max over the 91 real classes -> scores
        if (threadIdx.x < 32) {
            int r = threadIdx.x, gm = blockIdx.x * 32 + r;
            if (gm < M) {
                float s = -__builtin_inff();
                for (int j = 0; j < NCLS; ++j) {
                    float v = lds[r * N + j] + bias[j];
                    s = v > s ? v : s;
                }
                scores[gm] = s;
            }
        }
    }
}

// ---------------------------------------------------------------------------
// Kernel 4: coords_unact = h2 @ W3 + b3 + proposals   (N=4, scalar path)
// ---------------------------------------------------------------------------
__global__ __launch_bounds__(256) void coords_kernel(
        const bf16* __restrict__ h2, const float* __restrict__ W3,
        const float* __restrict__ b3, const float* __restrict__ opf,
        float* __restrict__ coords) {
    int tok = blockIdx.x * blockDim.x + threadIdx.x;
    if (tok >= M_TOT) return;
    float s0 = 0.f, s1 = 0.f, s2 = 0.f, s3 = 0.f;
    const bf16* h = h2 + (size_t)tok * CDIM;
    for (int j = 0; j < CDIM; ++j) {
        float v = (float)h[j];
        s0 += v * W3[j * 4 + 0];
        s1 += v * W3[j * 4 + 1];
        s2 += v * W3[j * 4 + 2];
        s3 += v * W3[j * 4 + 3];
    }
    float* o = coords + (size_t)tok * 4;
    const float* p = opf + (size_t)tok * 4;
    o[0] = s0 + b3[0] + p[0];
    o[1] = s1 + b3[1] + p[1];
    o[2] = s2 + b3[2] + p[2];
    o[3] = s3 + b3[3] + p[3];
}

// ---------------------------------------------------------------------------
// Kernel 5: per-batch top-20 (iterative block argmax over LDS-resident
// scores), gather + class recompute (160 rows), sigmoid/box, 20x20 NMS.
// ---------------------------------------------------------------------------
__global__ __launch_bounds__(256) void topnms_kernel(
        const float* __restrict__ scores, const bf16* __restrict__ outmem,
        const float* __restrict__ Wcls, const float* __restrict__ bcls,
        const float* __restrict__ coords, const float* __restrict__ WH,
        float* __restrict__ dout) {
    extern __shared__ float sm[];
    float* s_sc  = sm;                 // S_TOT
    float* r_val = sm + S_TOT;         // 256
    int*   r_idx = (int*)(sm + S_TOT + 256);
    __shared__ int   t_idx[TOPK];
    __shared__ float t_val[TOPK];
    __shared__ float bx[TOPK][4];
    __shared__ int   supp[TOPK];

    int b = blockIdx.x, tid = threadIdx.x;
    for (int i = tid; i < S_TOT; i += 256) s_sc[i] = scores[(size_t)b * S_TOT + i];
    __syncthreads();

    for (int p = 0; p < TOPK; ++p) {
        float bv = -__builtin_inff(); int bi = 0;
        for (int i = tid; i < S_TOT; i += 256)
            if (s_sc[i] > bv) { bv = s_sc[i]; bi = i; }
        r_val[tid] = bv; r_idx[tid] = bi;
        __syncthreads();
        for (int off = 128; off > 0; off >>= 1) {
            if (tid < off && r_val[tid + off] > r_val[tid]) {
                r_val[tid] = r_val[tid + off];
                r_idx[tid] = r_idx[tid + off];
            }
            __syncthreads();
        }
        if (tid == 0) {
            t_idx[p] = r_idx[0]; t_val[p] = r_val[0];
            s_sc[r_idx[0]] = -__builtin_inff();
        }
        __syncthreads();
    }

    // recompute class logits for the 20 selected rows (out[...,0:91])
    for (int kc = tid; kc < TOPK * NCLS; kc += 256) {
        int k = kc / NCLS, c = kc - k * NCLS;
        size_t tok = (size_t)b * S_TOT + t_idx[k];
        const bf16* row = outmem + tok * CDIM;
        float acc = 0.f;
        for (int j = 0; j < CDIM; ++j)
            acc += (float)row[j] * Wcls[(size_t)j * NCLS + c];
        dout[((size_t)b * TOPK + k) * (NCLS + 4) + c] = acc + bcls[c];
    }

    float* ref_out   = dout + (size_t)BATCH * TOPK * (NCLS + 4);
    float* keep_out  = ref_out + (size_t)BATCH * TOPK * 4;
    float* score_out = keep_out + (size_t)BATCH * TOPK;

    if (tid < TOPK) {
        int k = tid;
        size_t tok = (size_t)b * S_TOT + t_idx[k];
        float rp[4];
        for (int j = 0; j < 4; ++j) {
            float cu = coords[tok * 4 + j];
            dout[((size_t)b * TOPK + k) * (NCLS + 4) + NCLS + j] = cu;
            rp[j] = 1.0f / (1.0f + expf(-cu));
            ref_out[((size_t)b * TOPK + k) * 4 + j] = rp[j];
        }
        float cx = rp[0], cy = rp[1], w = rp[2], h = rp[3];
        bx[k][0] = truncf((cx - 0.5f * w) * WH[b * 4 + 0]);
        bx[k][1] = truncf((cy - 0.5f * h) * WH[b * 4 + 1]);
        bx[k][2] = truncf((cx + 0.5f * w) * WH[b * 4 + 2]);
        bx[k][3] = truncf((cy + 0.5f * h) * WH[b * 4 + 3]);
        score_out[(size_t)b * TOPK + k] = t_val[k];
        supp[k] = 0;
    }
    __syncthreads();

    if (tid == 0) {  // sequential NMS over already-descending order
        for (int i = 0; i < TOPK; ++i) {
            if (supp[i]) continue;
            float ai = (bx[i][2] - bx[i][0]) * (bx[i][3] - bx[i][1]);
            for (int j = i + 1; j < TOPK; ++j) {
                float xx1 = fmaxf(bx[i][0], bx[j][0]);
                float yy1 = fmaxf(bx[i][1], bx[j][1]);
                float xx2 = fminf(bx[i][2], bx[j][2]);
                float yy2 = fminf(bx[i][3], bx[j][3]);
                float inter = fmaxf(xx2 - xx1, 0.f) * fmaxf(yy2 - yy1, 0.f);
                float aj = (bx[j][2] - bx[j][0]) * (bx[j][3] - bx[j][1]);
                float iou = inter / (ai + aj - inter);
                if (iou > NMS_THR) supp[j] = 1;
            }
        }
        for (int k = 0; k < TOPK; ++k)
            keep_out[(size_t)b * TOPK + k] = supp[k] ? 0.0f : 1.0f;
    }
}

// ---------------------------------------------------------------------------
// Host launch
// ---------------------------------------------------------------------------
static inline void* carve(char*& p, size_t bytes) {
    void* r = (void*)p;
    p += (bytes + 255) & ~((size_t)255);
    return r;
}

extern "C" void kernel_launch(void* const* d_in, const int* in_sizes, int n_in,
                              void* d_out, int out_size, void* d_ws, size_t ws_size,
                              hipStream_t stream) {
    const float* memory = (const float*)d_in[0];
    const unsigned char* mask = (const unsigned char*)d_in[1];
    const float* WH   = (const float*)d_in[2];
    const float* Wenc = (const float*)d_in[3];
    const float* benc = (const float*)d_in[4];
    const float* lng  = (const float*)d_in[5];
    const float* lnb  = (const float*)d_in[6];
    const float* Wcls = (const float*)d_in[7];
    const float* bcls = (const float*)d_in[8];
    const float* W1   = (const float*)d_in[9];
    const float* b1   = (const float*)d_in[10];
    const float* W2   = (const float*)d_in[11];
    const float* b2   = (const float*)d_in[12];
    const float* W3   = (const float*)d_in[13];
    const float* b3   = (const float*)d_in[14];

    char* wp = (char*)d_ws;
    bf16* memA   = (bf16*)carve(wp, (size_t)M_TOT * CDIM * sizeof(bf16)); // reused as h2
    bf16* outmem = (bf16*)carve(wp, (size_t)M_TOT * CDIM * sizeof(bf16));
    bf16* h1     = (bf16*)carve(wp, (size_t)M_TOT * CDIM * sizeof(bf16));
    bf16* BpEnc  = (bf16*)carve(wp, (size_t)CDIM * CDIM * sizeof(bf16));
    bf16* BpW1   = (bf16*)carve(wp, (size_t)CDIM * CDIM * sizeof(bf16));
    bf16* BpW2   = (bf16*)carve(wp, (size_t)CDIM * CDIM * sizeof(bf16));
    bf16* BpCls  = (bf16*)carve(wp, (size_t)CDIM * NCLS_PAD * sizeof(bf16));
    float* opf    = (float*)carve(wp, (size_t)M_TOT * 4 * sizeof(float));
    float* coords = (float*)carve(wp, (size_t)M_TOT * 4 * sizeof(float));
    float* scores = (float*)carve(wp, (size_t)M_TOT * sizeof(float));
    float* vwh    = (float*)carve(wp, (size_t)BATCH * 4 * 2 * sizeof(float));
    bf16* h2 = memA;  // memory copy dead after GEMM1

    validwh_kernel<<<BATCH * 4, 32, 0, stream>>>(mask, vwh);
    proposals_kernel<<<(M_TOT + 7) / 8, 256, 0, stream>>>(memory, mask, vwh, memA, opf);

    packB_kernel<<<(16 * KSTEPS * 32 + 255) / 256, 256, 0, stream>>>(Wenc, BpEnc, CDIM, 16);
    packB_kernel<<<(16 * KSTEPS * 32 + 255) / 256, 256, 0, stream>>>(W1,   BpW1,  CDIM, 16);
    packB_kernel<<<(16 * KSTEPS * 32 + 255) / 256, 256, 0, stream>>>(W2,   BpW2,  CDIM, 16);
    packB_kernel<<<(8  * KSTEPS * 32 + 255) / 256, 256, 0, stream>>>(Wcls, BpCls, NCLS, 8);

    int gblocks = (M_TOT + 31) / 32;
    // out_mem = LN(memory @ W_enc + b_enc)
    wmma_gemm<16, 0><<<gblocks, 256, 0, stream>>>(memA, BpEnc, benc, lng, lnb,
                                                  outmem, nullptr, M_TOT);
    // scores = rowmax(out_mem @ W_cls + b_cls)   (N padded 91 -> 128)
    wmma_gemm<8, 2><<<gblocks, 256, 0, stream>>>(outmem, BpCls, bcls, nullptr, nullptr,
                                                 nullptr, scores, M_TOT);
    // h1 = relu(out_mem @ W1 + b1)
    wmma_gemm<16, 1><<<gblocks, 256, 0, stream>>>(outmem, BpW1, b1, nullptr, nullptr,
                                                  h1, nullptr, M_TOT);
    // h2 = relu(h1 @ W2 + b2)
    wmma_gemm<16, 1><<<gblocks, 256, 0, stream>>>(h1, BpW2, b2, nullptr, nullptr,
                                                  h2, nullptr, M_TOT);

    coords_kernel<<<(M_TOT + 255) / 256, 256, 0, stream>>>(h2, W3, b3, opf, coords);

    size_t smbytes = (size_t)(S_TOT + 256 + 256) * sizeof(float);
    topnms_kernel<<<BATCH, 256, smbytes, stream>>>(scores, outmem, Wcls, bcls,
                                                   coords, WH, (float*)d_out);
}
